// GPT2SelfAttention_64433099374967
// MI455X (gfx1250) — matrix operand order repaired and checked by reading, not compiled, not run on hardware
//
#include <hip/hip_runtime.h>

typedef __attribute__((ext_vector_type(16))) __bf16 v16bf;
typedef __attribute__((ext_vector_type(8)))  float  v8f;
typedef int v4i __attribute__((vector_size(16)));   // matches builtin's V4i param

union FragAB { uint4 u[2]; v16bf v; };

static constexpr int Bb = 4, Ss = 2048, Hh = 12, Dd = 64, NXc = 768;

// ---- gfx1250 async Global->LDS DMA path (guarded: falls back to sync copy) ----
#if defined(__gfx1250__) && __has_builtin(__builtin_amdgcn_global_load_async_to_lds_b128)
#define USE_ASYNC_LDS 1
#else
#define USE_ASYNC_LDS 0
#endif

__device__ __forceinline__ void cp16(void* lds, const void* g) {
#if USE_ASYNC_LDS
    // generic->AS1 / generic->AS3 via integer casts (LDS offset = low 32 bits)
    __builtin_amdgcn_global_load_async_to_lds_b128(
        (__attribute__((address_space(1))) v4i*)(unsigned long long)g,
        (__attribute__((address_space(3))) v4i*)(unsigned long long)(unsigned int)(unsigned long long)lds,
        0, 0);
#else
    *(uint4*)lds = *(const uint4*)g;
#endif
}

__device__ __forceinline__ void cp_fence() {
#if USE_ASYNC_LDS
#if __has_builtin(__builtin_amdgcn_s_wait_asynccnt)
    __builtin_amdgcn_s_wait_asynccnt(0);
#else
    asm volatile("s_wait_asynccnt 0" ::: "memory");
#endif
#endif
    __syncthreads();
}

__device__ __forceinline__ unsigned short f2bf(float f) {
    unsigned int u = __float_as_uint(f);
    u += 0x7fffu + ((u >> 16) & 1u);   // round-to-nearest-even
    return (unsigned short)(u >> 16);
}

// ---------------- conversion kernels ----------------
__global__ __launch_bounds__(256) void k_cvt(const float* __restrict__ in,
                                             unsigned short* __restrict__ out, int n) {
    int i = blockIdx.x * 256 + threadIdx.x;
    if (i < n) out[i] = f2bf(in[i]);
}

// in[rows][cols] fp32 -> out[cols][rows] bf16
__global__ __launch_bounds__(256) void k_cvt_t(const float* __restrict__ in,
                                               unsigned short* __restrict__ out,
                                               int rows, int cols) {
    int i = blockIdx.x * 256 + threadIdx.x;
    if (i < rows * cols) {
        int r = i / cols, c = i % cols;
        out[(size_t)c * rows + r] = f2bf(in[i]);
    }
}

// ---------------- generic bf16 WMMA GEMM ----------------
// C[M][N] = A[M][K] * B^T (Bt stored [N][K]) + bias[N]
// mode 0: scatter into Q [B,H,S,D], K [B,H,S,D], V^T [B,H,D,S] (bf16)
// mode 1: store fp32 to outF[m*ldo + n]
#define BM 128
#define BN 128
#define BK 32

__global__ __launch_bounds__(256) void k_gemm(
    const unsigned short* __restrict__ A,
    const unsigned short* __restrict__ Bt,
    const float* __restrict__ bias,
    int K, int mode,
    unsigned short* __restrict__ outQ,
    unsigned short* __restrict__ outK,
    unsigned short* __restrict__ outV,
    float* __restrict__ outF, int ldo)
{
    __shared__ unsigned short sA[2][BM * BK];   // [m][k]
    __shared__ unsigned short sB[2][BN * BK];   // [n][k]

    const int tid  = threadIdx.x;
    const int wid  = tid >> 5, lane = tid & 31;
    const int half = lane >> 4, ln = lane & 15;
    const int wm = (wid & 3) * 32, wn = (wid >> 2) * 64;

    const int rowA0 = blockIdx.y * BM;
    const int rowB0 = blockIdx.x * BN;

    const int lr = tid >> 1;          // 0..127
    const int lk = (tid & 1) * 16;    // 0 or 16 (k offset within tile)

    const unsigned short* gA = A  + (size_t)(rowA0 + lr) * K + lk;
    const unsigned short* gB = Bt + (size_t)(rowB0 + lr) * K + lk;

    v8f acc[2][4];
#pragma unroll
    for (int i = 0; i < 2; ++i)
#pragma unroll
        for (int j = 0; j < 4; ++j)
#pragma unroll
            for (int e = 0; e < 8; ++e) acc[i][j][e] = 0.f;

#define STAGE(kt, bufi) do {                                            \
        const unsigned short* pA_ = gA + (size_t)(kt) * BK;             \
        const unsigned short* pB_ = gB + (size_t)(kt) * BK;             \
        cp16(&sA[bufi][lr * BK + lk],     pA_);                         \
        cp16(&sA[bufi][lr * BK + lk + 8], pA_ + 8);                     \
        cp16(&sB[bufi][lr * BK + lk],     pB_);                         \
        cp16(&sB[bufi][lr * BK + lk + 8], pB_ + 8);                     \
    } while (0)

    STAGE(0, 0);
    cp_fence();

    const int kTiles = K / BK;
    for (int kt = 0; kt < kTiles; ++kt) {
        const int buf = kt & 1;
        if (kt + 1 < kTiles) STAGE(kt + 1, (kt + 1) & 1);

        FragAB fa[2], fb[4];
#pragma unroll
        for (int mt = 0; mt < 2; ++mt) {
            const unsigned short* p = &sA[buf][(wm + mt * 16 + ln) * BK];
            fa[mt].u[0] = *(const uint4*)(p + half * 16);       // K = half*16 .. +7
            fa[mt].u[1] = *(const uint4*)(p + half * 16 + 8);   // K = half*16+8 ..
        }
#pragma unroll
        for (int nt = 0; nt < 4; ++nt) {
            const unsigned short* p = &sB[buf][(wn + nt * 16 + ln) * BK + half * 16];
            fb[nt].u[0] = *(const uint4*)(p);                   // K = half*16 .. +15
            fb[nt].u[1] = *(const uint4*)(p + 8);
        }
#pragma unroll
        for (int mt = 0; mt < 2; ++mt)
#pragma unroll
            for (int nt = 0; nt < 4; ++nt)
                acc[mt][nt] = __builtin_amdgcn_wmma_f32_16x16x32_bf16(
                    false, fa[mt].v, false, fb[nt].v, (short)0, acc[mt][nt], false, false);

        if (kt + 1 < kTiles) cp_fence();
    }
#undef STAGE

    // epilogue: C row M = r + 8*half, col N = ln
    // BN (=128) divides NX (=768) => whole block lies in one q/k/v region
    const int which = rowB0 / NXc;              // block-uniform
    const int nb0   = rowB0 - which * NXc;      // region-local column base
#pragma unroll
    for (int mt = 0; mt < 2; ++mt) {
#pragma unroll
        for (int nt = 0; nt < 4; ++nt) {
            const int n  = rowB0 + wn + nt * 16 + ln;
            const float bv = bias[n];
            const int nn = nb0 + wn + nt * 16 + ln;   // < 768
            const int h = nn >> 6, d = nn & 63;
#pragma unroll
            for (int r = 0; r < 8; ++r) {
                const int m = rowA0 + wm + mt * 16 + r + 8 * half;
                const float val = acc[mt][nt][r] + bv;
                if (mode == 0) {
                    const int b = m >> 11, s = m & (Ss - 1);
                    const size_t bh = (size_t)(b * Hh + h);
                    const unsigned short bf = f2bf(val);
                    if (which == 0)      outQ[(bh * Ss + s) * Dd + d] = bf;
                    else if (which == 1) outK[(bh * Ss + s) * Dd + d] = bf;
                    else                 outV[(bh * Dd + d) * Ss + s] = bf;
                } else {
                    outF[(size_t)m * ldo + n] = val;
                }
            }
        }
    }
}

// ---------------- flash attention ----------------
// block = (qtile, h, b); 8 waves x 16 q-rows = 128 q rows per block
__global__ __launch_bounds__(256) void k_attn(
    const unsigned short* __restrict__ Q,
    const unsigned short* __restrict__ Kk,
    const unsigned short* __restrict__ Vt,   // [B,H,D,S]
    const unsigned char*  __restrict__ am,   // [B,S] bool
    unsigned short* __restrict__ Z)          // [B,S,NX] bf16
{
    __shared__ unsigned short sK[2][32 * 64];   // [key][d]
    __shared__ unsigned short sV[2][64 * 32];   // [d][key]  (V^T chunk)
    __shared__ unsigned short sP[8][16 * 32];   // per-wave P scratch [qrow][key]

    const int tid  = threadIdx.x;
    const int wid  = tid >> 5, lane = tid & 31;
    const int half = lane >> 4, ln = lane & 15;

    const int b = blockIdx.z, h = blockIdx.y;
    const int qbase = blockIdx.x * 128;
    const size_t bh = (size_t)(b * Hh + h);

    // resident Q fragments: rows qbase + wid*16 + ln, K-dim = D = 64 (2 frags of 32)
    FragAB fq[2];
    const unsigned short* qp = Q + (bh * Ss + qbase + wid * 16 + ln) * Dd;
#pragma unroll
    for (int f = 0; f < 2; ++f) {
        fq[f].u[0] = *(const uint4*)(qp + f * 32 + half * 16);
        fq[f].u[1] = *(const uint4*)(qp + f * 32 + half * 16 + 8);
    }

    float mrow[8], lrow[8];
    v8f o[4];
#pragma unroll
    for (int r = 0; r < 8; ++r) { mrow[r] = -3.0e30f; lrow[r] = 0.f; }
#pragma unroll
    for (int nt = 0; nt < 4; ++nt)
#pragma unroll
        for (int e = 0; e < 8; ++e) o[nt][e] = 0.f;

    const int nch   = (qbase + 128) / 32;    // causal: keys up to block's last q row
    const int qrow0 = qbase + wid * 16 + 8 * half;

    const int key = tid >> 3, part = (tid & 7) * 8;   // K-chunk staging split
    const int dd  = tid >> 2, p2  = (tid & 3) * 8;    // V-chunk staging split

#define STAGE_KV(c, bufi) do {                                                   \
        const int kc_ = (c) * 32;                                                \
        cp16(&sK[bufi][key * 64 + part], Kk + (bh * Ss + kc_ + key) * Dd + part);\
        cp16(&sV[bufi][dd * 32 + p2],    Vt + (bh * Dd + dd) * Ss + kc_ + p2);   \
    } while (0)

    STAGE_KV(0, 0);
    cp_fence();

    for (int c = 0; c < nch; ++c) {
        const int kc0 = c * 32;
        const int buf = c & 1;
        if (c + 1 < nch) STAGE_KV(c + 1, (c + 1) & 1);

        // S = Q (16x64) * K^T (64x32): two 16x16 key-tiles, K-dim in 2 steps of 32
        v8f sc[2];
#pragma unroll
        for (int nt = 0; nt < 2; ++nt)
#pragma unroll
            for (int e = 0; e < 8; ++e) sc[nt][e] = 0.f;
#pragma unroll
        for (int ks = 0; ks < 2; ++ks) {
#pragma unroll
            for (int nt = 0; nt < 2; ++nt) {
                FragAB fk;
                const unsigned short* p = &sK[buf][(nt * 16 + ln) * 64 + ks * 32 + half * 16];
                fk.u[0] = *(const uint4*)(p);
                fk.u[1] = *(const uint4*)(p + 8);
                sc[nt] = __builtin_amdgcn_wmma_f32_16x16x32_bf16(
                    false, fq[ks].v, false, fk.v, (short)0, sc[nt], false, false);
            }
        }

        const int col0 = kc0 + ln, col1 = kc0 + 16 + ln;
        const float am0 = am[b * Ss + col0] ? 0.f : -10000.f;
        const float am1 = am[b * Ss + col1] ? 0.f : -10000.f;

        float va[8], vb2[8], cm[8];
#pragma unroll
        for (int r = 0; r < 8; ++r) {
            const int q_i = qrow0 + r;
            va[r]  = ((col0 <= q_i) ? sc[0][r] * 0.125f : -10000.f) + am0;
            vb2[r] = ((col1 <= q_i) ? sc[1][r] * 0.125f : -10000.f) + am1;
            cm[r]  = fmaxf(va[r], vb2[r]);
        }
#pragma unroll
        for (int off = 1; off < 16; off <<= 1)
#pragma unroll
            for (int r = 0; r < 8; ++r)
                cm[r] = fmaxf(cm[r], __shfl_xor(cm[r], off, 32));

        float pa[8], pb[8], rs[8];
#pragma unroll
        for (int r = 0; r < 8; ++r) {
            const float mn    = fmaxf(mrow[r], cm[r]);
            const float alpha = __expf(mrow[r] - mn);
            pa[r] = __expf(va[r] - mn);
            pb[r] = __expf(vb2[r] - mn);
            rs[r] = pa[r] + pb[r];
            lrow[r] *= alpha;
            mrow[r]  = mn;
#pragma unroll
            for (int nt = 0; nt < 4; ++nt) o[nt][r] *= alpha;
        }
#pragma unroll
        for (int off = 1; off < 16; off <<= 1)
#pragma unroll
            for (int r = 0; r < 8; ++r)
                rs[r] += __shfl_xor(rs[r], off, 32);
#pragma unroll
        for (int r = 0; r < 8; ++r) lrow[r] += rs[r];

        // C-layout -> A-layout transpose of P through per-wave LDS scratch
#pragma unroll
        for (int r = 0; r < 8; ++r) {
            sP[wid][(r + 8 * half) * 32 + ln]      = f2bf(pa[r]);
            sP[wid][(r + 8 * half) * 32 + 16 + ln] = f2bf(pb[r]);
        }
        asm volatile("s_wait_dscnt 0" ::: "memory");

        FragAB fp;
        fp.u[0] = *(const uint4*)&sP[wid][ln * 32 + half * 16];
        fp.u[1] = *(const uint4*)&sP[wid][ln * 32 + half * 16 + 8];

        // O += P (16x32) * V (32x64): 4 d-tiles, B-frags from V^T chunk
#pragma unroll
        for (int nt = 0; nt < 4; ++nt) {
            FragAB fv;
            const unsigned short* p = &sV[buf][(nt * 16 + ln) * 32 + half * 16];
            fv.u[0] = *(const uint4*)(p);
            fv.u[1] = *(const uint4*)(p + 8);
            o[nt] = __builtin_amdgcn_wmma_f32_16x16x32_bf16(
                false, fp.v, false, fv.v, (short)0, o[nt], false, false);
        }

        if (c + 1 < nch) cp_fence();
    }
#undef STAGE_KV

    // epilogue: Z[(b*S+s)*NX + h*64 + d] = O / l
#pragma unroll
    for (int r = 0; r < 8; ++r) {
        const int s = qrow0 + r;
        const float inv = 1.0f / lrow[r];
#pragma unroll
        for (int nt = 0; nt < 4; ++nt)
            Z[((size_t)(b * Ss + s)) * NXc + h * 64 + nt * 16 + ln] =
                f2bf(o[nt][r] * inv);
    }
}

// ---------------- host launcher ----------------
extern "C" void kernel_launch(void* const* d_in, const int* in_sizes, int n_in,
                              void* d_out, int out_size, void* d_ws, size_t ws_size,
                              hipStream_t stream) {
    (void)in_sizes; (void)n_in; (void)out_size; (void)ws_size;
    const float*         x      = (const float*)d_in[0];
    const unsigned char* amask  = (const unsigned char*)d_in[1];
    const float*         W_attn = (const float*)d_in[2];
    const float*         b_attn = (const float*)d_in[3];
    const float*         W_proj = (const float*)d_in[4];
    const float*         b_proj = (const float*)d_in[5];
    float*               out    = (float*)d_out;

    char* ws = (char*)d_ws;
    size_t off = 0;
    auto alloc = [&](size_t bytes) -> char* {
        char* p = ws + off;
        off += (bytes + 255) & ~(size_t)255;
        return p;
    };
    const size_t MS = (size_t)Bb * Ss;  // 8192 rows
    unsigned short* xb   = (unsigned short*)alloc(MS * NXc * 2);
    unsigned short* wqkt = (unsigned short*)alloc((size_t)3 * NXc * NXc * 2);
    unsigned short* wpt  = (unsigned short*)alloc((size_t)NXc * NXc * 2);
    unsigned short* Qb   = (unsigned short*)alloc(MS * NXc * 2);
    unsigned short* Kb   = (unsigned short*)alloc(MS * NXc * 2);
    unsigned short* Vtb  = (unsigned short*)alloc(MS * NXc * 2);
    unsigned short* Zb   = (unsigned short*)alloc(MS * NXc * 2);

    {   // x -> bf16
        int n = (int)(MS * NXc);
        k_cvt<<<(n + 255) / 256, 256, 0, stream>>>(x, xb, n);
    }
    {   // W_attn [768][2304] -> [2304][768] bf16
        int n = NXc * 3 * NXc;
        k_cvt_t<<<(n + 255) / 256, 256, 0, stream>>>(W_attn, wqkt, NXc, 3 * NXc);
    }
    {   // W_proj [768][768] -> transposed bf16
        int n = NXc * NXc;
        k_cvt_t<<<(n + 255) / 256, 256, 0, stream>>>(W_proj, wpt, NXc, NXc);
    }

    // QKV GEMM: M=8192, N=2304, K=768
    k_gemm<<<dim3(3 * NXc / BN, MS / BM), 256, 0, stream>>>(
        xb, wqkt, b_attn, NXc, 0, Qb, Kb, Vtb, nullptr, 0);

    // flash attention: (qtile, h, b)
    k_attn<<<dim3(Ss / 128, Hh, Bb), 256, 0, stream>>>(Qb, Kb, Vtb, amask, Zb);

    // output projection: M=8192, N=768, K=768, fp32 out + bias
    k_gemm<<<dim3(NXc / BN, MS / BM), 256, 0, stream>>>(
        Zb, wpt, b_proj, NXc, 1, nullptr, nullptr, nullptr, out, NXc);
}
